// LSTMSimulator_64819646431332
// MI455X (gfx1250) — matrix operand order repaired and checked by reading, not compiled, not run on hardware
//
#include <hip/hip_runtime.h>
#include <hip/hip_bf16.h>

typedef _Float16 v16h __attribute__((ext_vector_type(16)));
typedef _Float16 h2   __attribute__((ext_vector_type(2)));
typedef float    v8f  __attribute__((ext_vector_type(8)));
typedef float    v2f  __attribute__((ext_vector_type(2)));

#define B_TOT 8192
#define T_LEN 2048
#define NH    32
#define WAVES_PER_BLK 4

// ---------------- fast activation primitives ----------------
__device__ __forceinline__ float fast_rcp(float x) {
#if __has_builtin(__builtin_amdgcn_rcpf)
    return __builtin_amdgcn_rcpf(x);
#else
    return 1.0f / x;
#endif
}
__device__ __forceinline__ float fast_exp2(float x) {
#if __has_builtin(__builtin_amdgcn_exp2f)
    return __builtin_amdgcn_exp2f(x);
#else
    return exp2f(x);
#endif
}

#if __has_builtin(__builtin_amdgcn_tanhf)
#define HAVE_TANHF 1
#endif

#ifdef HAVE_TANHF
// gate pre-activations arrive pre-scaled: p = 0.5*x for i/f/o, p = x for g
#define SCALE_IFO 0.5f
#define SCALE_G   1.0f
__device__ __forceinline__ float sigmoid_pre(float p) {           // sigmoid(x)=0.5+0.5*tanh(x/2)
    return __builtin_amdgcn_tanhf(p) * 0.5f + 0.5f;
}
__device__ __forceinline__ float tanh_pre(float p)  { return __builtin_amdgcn_tanhf(p); }
__device__ __forceinline__ float tanh_raw(float x)  { return __builtin_amdgcn_tanhf(x); }
#else
// gate pre-activations arrive pre-scaled: p = -log2e*x (i/f/o), p = -2log2e*x (g)
#define SCALE_IFO (-1.442695040888963f)
#define SCALE_G   (-2.885390081777927f)
__device__ __forceinline__ float sigmoid_pre(float p) {           // 1/(1+2^p)
    return fast_rcp(1.0f + fast_exp2(p));
}
__device__ __forceinline__ float tanh_pre(float p) {              // 2/(1+2^p)-1
    return fast_rcp(1.0f + fast_exp2(p)) * 2.0f - 1.0f;
}
__device__ __forceinline__ float tanh_raw(float x) {
    return fast_rcp(1.0f + fast_exp2(SCALE_G * x)) * 2.0f - 1.0f;
}
#endif

// K-permutation shared by the A reload (packed-pair LDS layout) and the B tiles.
__device__ __forceinline__ int sigma_k(int kp) {
    const int b    = kp >> 3;                          // K' block
    const int base = b << 2;                           // {0,4,8,12}
    return base + ((kp & 7) >> 1) + ((kp & 1) << 4);
}

#define WMMA32(A, B, C) __builtin_amdgcn_wmma_f32_16x16x32_f16(false, (A), false, (B), (short)0, (C), false, false)
#define WMMA4(A, B, C)  __builtin_amdgcn_wmma_f32_16x16x4_f32 (false, (A), false, (B), (short)0, (C), false, false)

// One wave owns a 16-row batch tile for all T steps.
// Gate tiles (16 N-cols each): i->{0,1}, f->{2,3}, g->{4,5}, o->{6,7}.
__global__ __launch_bounds__(WAVES_PER_BLK * 32)
void lstm_wmma_kernel(const float* __restrict__ input,
                      const float* __restrict__ W_ih,
                      const float* __restrict__ W_hh,
                      const float* __restrict__ b_ih,
                      const float* __restrict__ b_hh,
                      const float* __restrict__ W_lin,
                      const float* __restrict__ b_lin,
                      float* __restrict__ out)
{
    __shared__ __align__(64) h2 hbuf[WAVES_PER_BLK][16][16]; // packed (k, k+16) f16 pairs

    const int lane = threadIdx.x & 31;
    const int wave = threadIdx.x >> 5;
    const int lo   = lane & 15;
    const int hi   = lane >> 4;
    const int b0   = (blockIdx.x * WAVES_PER_BLK + wave) * 16;

    // ---- one-time: W_hh^T as 8 B-tiles (32x16 f16), K-permuted + gate-scaled ----
    v16h Bhh[8];
    #pragma unroll
    for (int n = 0; n < 8; ++n) {
        const float sc = (n == 4 || n == 5) ? SCALE_G : SCALE_IFO;
        const float* src = W_hh + (16 * n + lo) * NH;   // row of W_hh = column of B
        v16h b;
        #pragma unroll
        for (int i = 0; i < 16; ++i) b[i] = (_Float16)(sc * src[sigma_k(16 * hi + i)]);
        Bhh[n] = b;
    }
    // B_lin: only column N=0 nonzero = W_lin[k] (unscaled), same K permutation
    v16h Blin;
    #pragma unroll
    for (int i = 0; i < 16; ++i) Blin[i] = (_Float16)0.0f;
    if (lo == 0) {
        #pragma unroll
        for (int i = 0; i < 16; ++i) Blin[i] = (_Float16)W_lin[sigma_k(16 * hi + i)];
    }

    // ---- one-time: rank-1 (x,bias) operands for the K=4 f32 WMMA ----
    // B 4x16 f32 layout: V0 = row K=0 (lanes 0-15) / K=2 (lanes 16-31);
    //                    V1 = row K=1 / K=3.  Rows: K=0 -> W_ih (scaled),
    //                    K=1 -> fused bias (scaled), K=2,3 -> 0.
    v2f Bw[8];
    #pragma unroll
    for (int n = 0; n < 8; ++n) {
        const float sc = (n == 4 || n == 5) ? SCALE_G : SCALE_IFO;
        const int j = 16 * n + lo;
        v2f w;
        w[0] = (hi == 0) ? sc * W_ih[j] : 0.0f;
        w[1] = (hi == 0) ? sc * (b_ih[j] + b_hh[j]) : 0.0f;
        Bw[n] = w;
    }
    const float aone = (hi == 0) ? 1.0f : 0.0f;        // A column K=1 (all-ones)

    // persistent C tiles (WMMA reads C without modifying it)
    v8f zeroC;
    #pragma unroll
    for (int r = 0; r < 8; ++r) zeroC[r] = 0.0f;
    const float blin = b_lin[0];
    v8f yaC;
    #pragma unroll
    for (int r = 0; r < 8; ++r) yaC[r] = blin;

    // state: c in C/D layout (cst[g][r] = c[M=r+8hi][k=lo+16g]); h in A layout f16
    float cst[2][8];
    #pragma unroll
    for (int g = 0; g < 2; ++g)
        #pragma unroll
        for (int r = 0; r < 8; ++r) cst[g][r] = 0.0f;
    v16h hA;
    #pragma unroll
    for (int i = 0; i < 16; ++i) hA[i] = (_Float16)0.0f;

    const float* xrow = input + (size_t)(b0 + lo) * T_LEN;

    // software-pipelined x: Ax for step t is built from the load issued at step t-1,
    // so the global load latency hides behind a full step of WMMA/tanh work.
    v2f Ax;
    Ax[0] = (hi == 0) ? xrow[0] : 0.0f;
    Ax[1] = aone;

    for (int t = 0; t < T_LEN; ++t) {
        const float xnext = xrow[(t + 1) & (T_LEN - 1)];   // in-bounds clamp (T pow2)
        __builtin_prefetch(&xrow[t + 16], 0, 3);           // WGP-scope prefetch

        _Float16 h0h[8];                               // group-0 h halves, packed at group 1
        #pragma unroll
        for (int g = 0; g < 2; ++g) {
            // gate_pre = x (x) W_ih + bias  (rank-1 via K=4 f32 WMMA, C=0 inline)
            v8f pi = WMMA4(Ax, Bw[0 + g], zeroC);
            v8f pf = WMMA4(Ax, Bw[2 + g], zeroC);
            v8f pg = WMMA4(Ax, Bw[4 + g], zeroC);
            v8f po = WMMA4(Ax, Bw[6 + g], zeroC);
            // gate = h @ W_hh^T + gate_pre
            v8f ti = WMMA32(hA, Bhh[0 + g], pi);
            v8f tf = WMMA32(hA, Bhh[2 + g], pf);
            v8f tg = WMMA32(hA, Bhh[4 + g], pg);
            v8f to = WMMA32(hA, Bhh[6 + g], po);
            #pragma unroll
            for (int r = 0; r < 8; ++r) {
                const float iv = sigmoid_pre(ti[r]);
                const float fv = sigmoid_pre(tf[r]);
                const float gv = tanh_pre(tg[r]);
                const float ov = sigmoid_pre(to[r]);
                const float cn = fv * cst[g][r] + iv * gv;
                cst[g][r] = cn;
                const float hn = ov * tanh_raw(cn);
                if (g == 0) {
                    h0h[r] = (_Float16)hn;
                } else {                               // pack (k, k+16) -> one ds_store_b32
                    h2 v; v[0] = h0h[r]; v[1] = (_Float16)hn;
                    hbuf[wave][r + hi * 8][lo] = v;
                }
            }
        }

        // A reload: one contiguous 32B v16h per lane (2x ds_load_b128), K order = sigma
        hA = *(const v16h*)&hbuf[wave][lo][8 * hi];

        // y = h_new @ W_lin^T + b_lin via WMMA (N=0 column), C = persistent bias splat
        v8f ya = WMMA32(hA, Blin, yaC);
        if (lo == 0) {
            #pragma unroll
            for (int r = 0; r < 8; ++r)
                out[(size_t)(b0 + hi * 8 + r) * T_LEN + t] = ya[r];
        }

        // consume the pipelined load only here -> s_wait_loadcnt lands at loop bottom
        Ax[0] = (hi == 0) ? xnext : 0.0f;
    }
}

extern "C" void kernel_launch(void* const* d_in, const int* in_sizes, int n_in,
                              void* d_out, int out_size, void* d_ws, size_t ws_size,
                              hipStream_t stream) {
    (void)in_sizes; (void)n_in; (void)out_size; (void)d_ws; (void)ws_size;
    const float* input = (const float*)d_in[0];
    const float* W_ih  = (const float*)d_in[1];
    const float* W_hh  = (const float*)d_in[2];
    const float* b_ih  = (const float*)d_in[3];
    const float* b_hh  = (const float*)d_in[4];
    const float* W_lin = (const float*)d_in[5];
    const float* b_lin = (const float*)d_in[6];
    float* out = (float*)d_out;

    dim3 grid(B_TOT / (16 * WAVES_PER_BLK));   // 128 blocks
    dim3 block(32 * WAVES_PER_BLK);            // 4 waves/block, 512 waves total
    hipLaunchKernelGGL(lstm_wmma_kernel, grid, block, 0, stream,
                       input, W_ih, W_hh, b_ih, b_hh, W_lin, b_lin, out);
}
